// GroupedAttention_66468913873288
// MI455X (gfx1250) — compile-verified, matrix operand order checked by reading, not run on hardware
//
#include <hip/hip_runtime.h>
#include <hip/hip_bf16.h>

#define DEV_INLINE __device__ __forceinline__

typedef __bf16 bf16_t;
typedef bf16_t v8bf  __attribute__((ext_vector_type(8)));
typedef bf16_t v16bf __attribute__((ext_vector_type(16)));
typedef float  v8f   __attribute__((ext_vector_type(8)));

constexpr int B_   = 2;
constexpr int T_   = 2048;
constexpr int H_   = 16;
constexpr int G_   = 4;
constexpr int DQ_  = 32;    // query/key head dim
constexpr int DV_  = 64;    // value head dim
constexpr int DIN  = 1024;
constexpr int DOUT = 1024;
constexpr int ROWS = B_ * T_;      // 4096
constexpr int NQ   = H_ * DQ_;     // 512
constexpr int NK   = G_ * DQ_;     // 128
constexpr int NV   = G_ * DV_;     // 256

// ---------------------------------------------------------------------------
// WMMA helper (CDNA5 wave32, V_WMMA_F32_16X16X32_BF16)
// ---------------------------------------------------------------------------
DEV_INLINE v8f wmma_bf16(v16bf a, v16bf b, v8f c) {
  return __builtin_amdgcn_wmma_f32_16x16x32_bf16(false, a, false, b,
                                                 (short)0, c, false, false);
}

// 16-bit A/B fragment from a 16x32 row-major tile (ld = row stride, elems).
// lane<16  : row = lane,    K v0..3 = 0..7,  v4..7 = 16..23
// lane>=16 : row = lane-16, K v0..3 = 8..15, v4..7 = 24..31
DEV_INLINE v16bf load_frag(const bf16_t* base, int ld) {
  const int lane = (int)(threadIdx.x & 31u);
  const bf16_t* p = base + (size_t)(lane & 15) * (size_t)ld + ((lane >> 4) << 3);
  v8bf lo = *reinterpret_cast<const v8bf*>(p);
  v8bf hi = *reinterpret_cast<const v8bf*>(p + 16);
  return __builtin_shufflevector(lo, hi, 0, 1, 2, 3, 4, 5, 6, 7,
                                 8, 9, 10, 11, 12, 13, 14, 15);
}

// ---------------------------------------------------------------------------
// gfx1250 async global->LDS staging (ASYNCcnt path)
// Copies a 128-row x 32-col bf16 tile (8 KB) with 256 threads:
// 512 x 16B chunks -> 2 async b128 instructions per wave-slot per thread.
// ---------------------------------------------------------------------------
DEV_INLINE void stage_tile_async(const bf16_t* __restrict__ g, int ldg,
                                 bf16_t* l) {
  const int tid = (int)threadIdx.x;
#pragma unroll
  for (int c = tid; c < 512; c += 256) {
    const int row = c >> 2;             // 4 chunks of 8 bf16 per 32-elem row
    const int col = (c & 3) << 3;
    const bf16_t* gp = g + (size_t)row * ldg + col;
    const unsigned int lp =
        (unsigned int)(size_t)(const void*)(l + row * 32 + col);
    asm volatile("global_load_async_to_lds_b128 %0, %1, off"
                 :
                 : "v"(lp), "v"((unsigned long long)(size_t)gp)
                 : "memory");
  }
}

DEV_INLINE void wait_async_le4() { asm volatile("s_wait_asynccnt 0x4" ::: "memory"); }
DEV_INLINE void wait_async_0()   { asm volatile("s_wait_asynccnt 0x0" ::: "memory"); }

// ---------------------------------------------------------------------------
// Conversion kernels
// ---------------------------------------------------------------------------
__global__ void ga_f32_to_bf16(const float* __restrict__ src,
                               bf16_t* __restrict__ dst, int n) {
  int i = blockIdx.x * blockDim.x + threadIdx.x;
  if (i < n) dst[i] = (bf16_t)src[i];
}

// W (K x N, row-major f32)  ->  WT (N x K, row-major bf16)
__global__ void ga_transpose_bf16(const float* __restrict__ W,
                                  bf16_t* __restrict__ WT, int K, int N) {
  int i = blockIdx.x * blockDim.x + threadIdx.x;
  if (i < K * N) {
    int k = i / N;
    int n = i - k * N;
    WT[(size_t)n * K + k] = (bf16_t)W[i];
  }
}

// ---------------------------------------------------------------------------
// Double-buffered LDS-tiled bf16 WMMA GEMM:  C[M,N] = A[M,K] * BT[N,K]^T
// Block = 256 threads (8 waves); block tile 128M x 128N; K-step 32.
// Wave tile = 32M x 64N -> 2 A-frags x 4 B-frags -> 8 WMMA per K-step.
// Tiles staged with global_load_async_to_lds_b128, prefetch depth 1.
// ---------------------------------------------------------------------------
template <bool F32OUT>
__global__ __launch_bounds__(256)
void ga_gemm_async(const bf16_t* __restrict__ A, const bf16_t* __restrict__ BT,
                   const float* __restrict__ bias, void* __restrict__ Cout,
                   int M, int N, int K) {
  __shared__ __align__(16) bf16_t sA[2][128 * 32];
  __shared__ __align__(16) bf16_t sB[2][128 * 32];
  const int wave = (int)(threadIdx.x >> 5);
  const int lane = (int)(threadIdx.x & 31u);
  const int mBlock = blockIdx.y * 128;
  const int nBlock = blockIdx.x * 128;
  const int mW = (wave >> 1) * 32;   // 4 waves along M
  const int nW = (wave & 1) * 64;    // 2 waves along N

  v8f acc[2][4] = {};
  const int KT = K >> 5;

  stage_tile_async(A + (size_t)mBlock * K, K, sA[0]);
  stage_tile_async(BT + (size_t)nBlock * K, K, sB[0]);

  for (int kt = 0; kt < KT; ++kt) {
    const int cur = kt & 1;
    if (kt + 1 < KT) {
      stage_tile_async(A + (size_t)mBlock * K + (size_t)(kt + 1) * 32, K,
                       sA[cur ^ 1]);
      stage_tile_async(BT + (size_t)nBlock * K + (size_t)(kt + 1) * 32, K,
                       sB[cur ^ 1]);
      wait_async_le4();   // tile kt complete; 4 prefetch ops still in flight
    } else {
      wait_async_0();
    }
    __syncthreads();

    v16bf aF[2], bF[4];
#pragma unroll
    for (int i = 0; i < 2; ++i)
      aF[i] = load_frag(sA[cur] + (size_t)(mW + i * 16) * 32, 32);
#pragma unroll
    for (int t = 0; t < 4; ++t)
      bF[t] = load_frag(sB[cur] + (size_t)(nW + t * 16) * 32, 32);
#pragma unroll
    for (int i = 0; i < 2; ++i)
#pragma unroll
      for (int t = 0; t < 4; ++t)
        acc[i][t] = wmma_bf16(aF[i], bF[t], acc[i][t]);

    __syncthreads();  // all waves done reading before buffer is re-staged
  }

  const int n = lane & 15, mo = (lane >> 4) << 3;
#pragma unroll
  for (int i = 0; i < 2; ++i)
#pragma unroll
    for (int t = 0; t < 4; ++t)
#pragma unroll
      for (int r = 0; r < 8; ++r) {
        const size_t row = (size_t)(mBlock + mW + i * 16 + mo + r);
        const int col = nBlock + nW + t * 16 + n;
        if constexpr (F32OUT)
          ((float*)Cout)[row * N + col] = acc[i][t][r] + bias[col];
        else
          ((bf16_t*)Cout)[row * N + col] = (bf16_t)acc[i][t][r];
      }
}

// ---------------------------------------------------------------------------
// Flash attention: one wave per (b, h, 16-query tile).
// Grid: x = T/16 = 128, y = B*H = 32. Block = 32 threads (1 wave).
// DQ=32 -> exactly one v_wmma_f32_16x16x32_bf16 per 16x16 score tile.
// ---------------------------------------------------------------------------
__global__ __launch_bounds__(32)
void ga_attention(const bf16_t* __restrict__ qb,   // (B*T, 512)
                  const bf16_t* __restrict__ kbuf, // (B*T, 128)
                  const bf16_t* __restrict__ vbuf, // (B*T, 256)
                  bf16_t* __restrict__ ctx) {      // (B*T, 1024)
  __shared__ __align__(16) float  sS[16 * 32];
  __shared__ __align__(16) bf16_t sP[16 * 32];
  __shared__ __align__(16) bf16_t sVt[64 * 32];   // [dim][key]
  __shared__ float sM[16], sL[16], sA[16];

  const int lane = (int)threadIdx.x;
  const int qt = blockIdx.x;
  const int bh = blockIdx.y;
  const int b = bh >> 4, h = bh & 15, kvh = h & 3;
  const int qBase = qt * 16;
  const float slope = exp2f(-0.5f * (float)(h + 1));   // 2^(-8(h+1)/16)
  const float rsq = 0.17677669529663687f;              // 1/sqrt(32)

  const bf16_t* qRow = qb + ((size_t)b * T_ + qBase) * NQ + h * DQ_;
  const v16bf aq = load_frag(qRow, NQ);                // K = DQ = 32 exactly

  if (lane < 16) { sM[lane] = -__builtin_inff(); sL[lane] = 0.0f; }
  __syncthreads();

  v8f acc[4] = {};
  const int n = lane & 15, mo = (lane >> 4) << 3;
  const int nkb = (qBase + 16 + 31) >> 5;              // causal key-block count

  for (int kb = 0; kb < nkb; ++kb) {
    const int kPos = kb * 32;
    // ---- scores: one WMMA per 16-key half-block ----
    const bf16_t* kRow = kbuf + ((size_t)b * T_ + kPos) * NK + kvh * DQ_;
    v16bf bk0 = load_frag(kRow, NK);
    v16bf bk1 = load_frag(kRow + (size_t)16 * NK, NK);
    v8f z = {};
    v8f s0 = wmma_bf16(aq, bk0, z);
    v8f s1 = wmma_bf16(aq, bk1, z);

    __syncthreads();  // previous iteration's sP / sVt fully consumed
    // mask + alibi + scale in C-fragment registers, spill tile to LDS
#pragma unroll
    for (int r = 0; r < 8; ++r) {
      const int m = mo + r;
      const int qp = qBase + m;
      const int k0p = kPos + n, k1p = k0p + 16;
      sS[m * 32 + n] = (k0p > qp) ? -__builtin_inff()
                                  : (s0[r] + slope * (float)(k0p - qp)) * rsq;
      sS[m * 32 + n + 16] = (k1p > qp) ? -__builtin_inff()
                                       : (s1[r] + slope * (float)(k1p - qp)) * rsq;
    }
    // stage V block transposed: sVt[d][k] = V[kPos+k, kvh*64+d]
    const bf16_t* vRow = vbuf + ((size_t)b * T_ + kPos) * NV + kvh * DV_;
    for (int idx = lane; idx < 64 * 32; idx += 32) {
      const int kk = idx >> 6, d = idx & 63;
      sVt[d * 32 + kk] = vRow[(size_t)kk * NV + d];
    }
    __syncthreads();

    // ---- online softmax (lanes 0..15 own one query row each) ----
    if (lane < 16) {
      const int m = lane;
      float mOld = sM[m], rmax = mOld;
      for (int j = 0; j < 32; ++j) rmax = fmaxf(rmax, sS[m * 32 + j]);
      const float alpha = __expf(mOld - rmax);
      float sum = 0.0f;
      for (int j = 0; j < 32; ++j) {
        const float p = __expf(sS[m * 32 + j] - rmax);
        sP[m * 32 + j] = (bf16_t)p;
        sum += p;
      }
      sM[m] = rmax;
      sL[m] = sL[m] * alpha + sum;
      sA[m] = alpha;
    }
    __syncthreads();

    // ---- rescale running context, then acc += P @ V ----
#pragma unroll
    for (int t = 0; t < 4; ++t)
#pragma unroll
      for (int r = 0; r < 8; ++r) acc[t][r] *= sA[mo + r];

    const v16bf aP = load_frag(sP, 32);                // P: 16x32 bf16
#pragma unroll
    for (int t = 0; t < 4; ++t) {
      v16bf bv = load_frag(sVt + (size_t)(t * 16) * 32, 32);
      acc[t] = wmma_bf16(aP, bv, acc[t]);
    }
  }

  // ---- finalize: divide by row sums, store ctx tile ----
  __syncthreads();
  if (lane < 16) sA[lane] = 1.0f / sL[lane];
  __syncthreads();
#pragma unroll
  for (int t = 0; t < 4; ++t)
#pragma unroll
    for (int r = 0; r < 8; ++r) {
      const float o = acc[t][r] * sA[mo + r];
      ctx[((size_t)b * T_ + qBase + mo + r) * DOUT + h * DV_ + t * 16 + n] =
          (bf16_t)o;
    }
}

// ---------------------------------------------------------------------------
// Host launcher
// ---------------------------------------------------------------------------
extern "C" void kernel_launch(void* const* d_in, const int* in_sizes, int n_in,
                              void* d_out, int out_size, void* d_ws,
                              size_t ws_size, hipStream_t stream) {
  (void)in_sizes; (void)n_in; (void)out_size; (void)ws_size;
  const float* x  = (const float*)d_in[0];
  const float* Wq = (const float*)d_in[1];
  const float* Wk = (const float*)d_in[2];
  const float* Wv = (const float*)d_in[3];
  const float* Wp = (const float*)d_in[4];
  const float* bp = (const float*)d_in[5];
  float* out = (float*)d_out;

  // workspace layout (bf16 buffers, 256B aligned)
  char* ws = (char*)d_ws;
  size_t off = 0;
  auto take = [&](size_t bytes) {
    char* p = ws + off;
    off += (bytes + 255) & ~(size_t)255;
    return p;
  };
  bf16_t* x_bf = (bf16_t*)take((size_t)ROWS * DIN  * 2);
  bf16_t* WqT  = (bf16_t*)take((size_t)NQ   * DIN  * 2);
  bf16_t* WkT  = (bf16_t*)take((size_t)NK   * DIN  * 2);
  bf16_t* WvT  = (bf16_t*)take((size_t)NV   * DIN  * 2);
  bf16_t* WpT  = (bf16_t*)take((size_t)DOUT * DOUT * 2);
  bf16_t* q_bf = (bf16_t*)take((size_t)ROWS * NQ   * 2);
  bf16_t* k_bf = (bf16_t*)take((size_t)ROWS * NK   * 2);
  bf16_t* v_bf = (bf16_t*)take((size_t)ROWS * NV   * 2);
  bf16_t* c_bf = (bf16_t*)take((size_t)ROWS * DOUT * 2);

  const int nx = ROWS * DIN;
  ga_f32_to_bf16<<<(nx + 255) / 256, 256, 0, stream>>>(x, x_bf, nx);
  ga_transpose_bf16<<<(DIN * NQ + 255) / 256, 256, 0, stream>>>(Wq, WqT, DIN, NQ);
  ga_transpose_bf16<<<(DIN * NK + 255) / 256, 256, 0, stream>>>(Wk, WkT, DIN, NK);
  ga_transpose_bf16<<<(DIN * NV + 255) / 256, 256, 0, stream>>>(Wv, WvT, DIN, NV);
  ga_transpose_bf16<<<(DOUT * DOUT + 255) / 256, 256, 0, stream>>>(Wp, WpT, DOUT, DOUT);

  // QKV projections (bf16 out), M = 4096 rows, block tile 128x128
  dim3 blk(256);
  ga_gemm_async<false><<<dim3(NQ / 128, ROWS / 128), blk, 0, stream>>>(
      x_bf, WqT, nullptr, q_bf, ROWS, NQ, DIN);
  ga_gemm_async<false><<<dim3(NK / 128, ROWS / 128), blk, 0, stream>>>(
      x_bf, WkT, nullptr, k_bf, ROWS, NK, DIN);
  ga_gemm_async<false><<<dim3(NV / 128, ROWS / 128), blk, 0, stream>>>(
      x_bf, WvT, nullptr, v_bf, ROWS, NV, DIN);

  // flash attention: 128 q-tiles x (B*H)=32
  ga_attention<<<dim3(T_ / 16, B_ * H_), dim3(32), 0, stream>>>(
      q_bf, k_bf, v_bf, c_bf);

  // output projection + bias (fp32 out)
  ga_gemm_async<true><<<dim3(DOUT / 128, ROWS / 128), blk, 0, stream>>>(
      c_bf, WpT, bp, out, ROWS, DOUT, DIN);
}
